// JointBlock_86560770884001
// MI455X (gfx1250) — matrix operand order repaired and checked by reading, not compiled, not run on hardware
//
#include <hip/hip_runtime.h>
#include <hip/hip_bf16.h>

// ---------------------------------------------------------------------------
// Types for CDNA5 WMMA / TDM
// ---------------------------------------------------------------------------
typedef __attribute__((ext_vector_type(16))) __bf16 v16bf;
typedef __attribute__((ext_vector_type(8)))  __bf16 v8bf;
typedef __attribute__((ext_vector_type(8)))  float  v8f;
typedef __attribute__((ext_vector_type(4)))  unsigned v4u;
typedef __attribute__((ext_vector_type(8)))  unsigned v8u;

#define B_   16
#define NL_  512
#define NC_  1024
#define C_   512
#define HEADS_ 8
#define HD_  64
#define NPAD 576   // 515 padded up to multiple of 64 (zero-filled)
#define KCHUNK 128 // TDM-staged K-chunk rows per double buffer

__device__ __forceinline__ float gelu_f(float x) {
    return 0.5f * x * (1.0f + erff(x * 0.70710678118654752f));
}

// A fragment: 16x32 bf16, A points at element (m0, k0) of row-major matrix.
__device__ __forceinline__ v16bf load_a_frag(const __bf16* A, int lda, int lane) {
    int m = lane & 15, hh = lane >> 4;
    const __bf16* p = A + (size_t)m * lda + 8 * hh;
    v8bf lo = *(const v8bf*)p;
    v8bf hi = *(const v8bf*)(p + 16);
    v16bf r;
#pragma unroll
    for (int e = 0; e < 8; ++e) { r[e] = lo[e]; r[e + 8] = hi[e]; }
    return r;
}

// B fragment: 32x16 bf16, Bm points at element (k0, n0); lane l = column n=l%16, k = 8h+e / 16+8h+e.
__device__ __forceinline__ v16bf load_b_frag(const __bf16* Bm, int ldb, int lane) {
    int n = lane & 15, hh = lane >> 4;
    const __bf16* p = Bm + n;
    v16bf r;
#pragma unroll
    for (int e = 0; e < 8; ++e) { r[e] = p[(size_t)(8 * hh + e) * ldb]; r[e + 8] = p[(size_t)(16 + 8 * hh + e) * ldb]; }
    return r;
}

// ---------------------------------------------------------------------------
// LayerNorm: f32 in -> bf16 out. One block (128 thr) per row, cols == 512.
// ---------------------------------------------------------------------------
__global__ void ln_kernel(const float* __restrict__ x, const float* __restrict__ w,
                          const float* __restrict__ b, __bf16* __restrict__ out, int cols) {
    int row = blockIdx.x, tid = threadIdx.x;
    const float* xr = x + (size_t)row * cols;
    float vals[4], s = 0.f, s2 = 0.f;
#pragma unroll
    for (int i = 0; i < 4; ++i) {
        float v = xr[tid + 128 * i];
        vals[i] = v; s += v; s2 += v * v;
    }
    __shared__ float r1[128], r2[128];
    r1[tid] = s; r2[tid] = s2; __syncthreads();
    for (int off = 64; off > 0; off >>= 1) {
        if (tid < off) { r1[tid] += r1[tid + off]; r2[tid] += r2[tid + off]; }
        __syncthreads();
    }
    float mean = r1[0] / (float)cols;
    float var  = r2[0] / (float)cols - mean * mean;
    float inv  = rsqrtf(var + 1e-5f);
#pragma unroll
    for (int i = 0; i < 4; ++i) {
        int c = tid + 128 * i;
        out[(size_t)row * cols + c] = (__bf16)((vals[i] - mean) * inv * w[c] + b[c]);
    }
}

// ---------------------------------------------------------------------------
// Weight conversions
// ---------------------------------------------------------------------------
__global__ void cvt_kernel(const float* __restrict__ s, __bf16* __restrict__ d, int n) {
    int i = blockIdx.x * 256 + threadIdx.x;
    if (i < n) d[i] = (__bf16)s[i];
}
__global__ void pad_wf_kernel(const float* __restrict__ s, __bf16* __restrict__ d) {
    int i = blockIdx.x * 256 + threadIdx.x;
    if (i >= 512 * NPAD) return;
    int row = i / NPAD, col = i % NPAD;
    d[i] = (col < 515) ? (__bf16)s[row * 515 + col] : (__bf16)0.0f;
}
__global__ void pad_bias_kernel(const float* __restrict__ s, float* __restrict__ d) {
    int i = blockIdx.x * 256 + threadIdx.x;
    if (i < NPAD) d[i] = (i < 515) ? s[i] : 0.f;
}

// ---------------------------------------------------------------------------
// Register-blocked bf16 WMMA GEMM with TDM-staged B panel.
// Block = 128 threads = 4 waves. Block tile: M=64, N=64.
// Wave w computes rows [blockIdx.y*64 + w*16, +16) x 64 cols (4 accumulators).
// B panel chunks (KCHUNK x 64 bf16 = 16 KB) are DMA'd into LDS by the Tensor
// Data Mover (tensor_load_to_lds, 2D descriptor) and double-buffered with
// s_wait_tensorcnt; A fragments stream from L2 via global_load_b128.
// Requires: M % 64 == 0, N % 64 == 0, K % KCHUNK == 0.
// ---------------------------------------------------------------------------
__global__ void gemm_bf16_kernel(const __bf16* __restrict__ A, const __bf16* __restrict__ Bm,
                                 const float* __restrict__ bias, const float* __restrict__ residual,
                                 float* __restrict__ outF, __bf16* __restrict__ outB,
                                 int M, int N, int K, int lda, int ldb, int ldc,
                                 int gelu_flag, int kT_flag) {
    __shared__ alignas(16) __bf16 Bs[2][KCHUNK * 64];
    int lane = threadIdx.x & 31;
    int wave = threadIdx.x >> 5;
    int n0 = blockIdx.x * 64;
    int m0 = blockIdx.y * 64 + wave * 16;
    int nchunks = K / KCHUNK;

    unsigned long long gbase = (unsigned long long)(const void*)(Bm + (size_t)n0);
    unsigned ldsa[2];
    ldsa[0] = (unsigned)(unsigned long long)(uintptr_t)&Bs[0][0];
    ldsa[1] = (unsigned)(unsigned long long)(uintptr_t)&Bs[1][0];

    // Build D# groups 0/1 (ISA 8.3/8.4) and issue TENSOR_LOAD_TO_LDS (2D form).
    auto issue_tdm = [&](int chunk, unsigned ldsaddr) {
        unsigned long long ga = gbase + (unsigned long long)chunk * KCHUNK * (unsigned long long)ldb * 2ull;
        v4u g0;
        g0[0] = 1u;                                  // count=1, user descriptor
        g0[1] = ldsaddr;                             // lds_addr (bytes)
        g0[2] = (unsigned)(ga & 0xffffffffu);        // global_addr[31:0]
        g0[3] = (unsigned)(ga >> 32) | (2u << 30);   // global_addr[56:32] | type=2
        unsigned td0 = (unsigned)ldb;                // tensor_dim0 (elements)
        unsigned td1 = (unsigned)K;                  // tensor_dim1 (rows)
        v8u g1;
        g1[0] = (1u << 16);                          // wg_mask=0 | data_size=1 (2B)
        g1[1] = (td0 & 0xffffu) << 16;               // tensor_dim0[15:0]
        g1[2] = ((td0 >> 16) & 0xffffu) | ((td1 & 0xffffu) << 16);
        g1[3] = ((td1 >> 16) & 0xffffu) | (64u << 16);          // tile_dim0 = 64
        g1[4] = (unsigned)KCHUNK;                    // tile_dim1 = KCHUNK, tile_dim2 = 0
        g1[5] = (unsigned)ldb;                       // tensor_dim0_stride[31:0]
        g1[6] = 0u;                                  // stride0 hi | stride1 lo
        g1[7] = 0u;
        asm volatile("tensor_load_to_lds %0, %1" :: "s"(g0), "s"(g1) : "memory");
    };

    if (wave == 0) issue_tdm(0, ldsa[0]);

    v8f acc0 = {}, acc1 = {}, acc2 = {}, acc3 = {};
    for (int chunk = 0; chunk < nchunks; ++chunk) {
        if (wave == 0) {
            if (chunk + 1 < nchunks) {
                issue_tdm(chunk + 1, ldsa[(chunk + 1) & 1]);
                __builtin_amdgcn_s_wait_tensorcnt(1);   // oldest (current chunk) done
            } else {
                __builtin_amdgcn_s_wait_tensorcnt(0);   // last chunk done
            }
        }
        __syncthreads();
        const __bf16* bs = &Bs[chunk & 1][0];
        int kbase = chunk * KCHUNK;
#pragma unroll
        for (int ks = 0; ks < KCHUNK; ks += 32) {
            v16bf af = load_a_frag(A + (size_t)m0 * lda + kbase + ks, lda, lane);
            v16bf b0 = load_b_frag(bs + ks * 64 +  0, 64, lane);
            v16bf b1 = load_b_frag(bs + ks * 64 + 16, 64, lane);
            v16bf b2 = load_b_frag(bs + ks * 64 + 32, 64, lane);
            v16bf b3 = load_b_frag(bs + ks * 64 + 48, 64, lane);
            acc0 = __builtin_amdgcn_wmma_f32_16x16x32_bf16(false, af, false, b0, (short)0, acc0, false, false);
            acc1 = __builtin_amdgcn_wmma_f32_16x16x32_bf16(false, af, false, b1, (short)0, acc1, false, false);
            acc2 = __builtin_amdgcn_wmma_f32_16x16x32_bf16(false, af, false, b2, (short)0, acc2, false, false);
            acc3 = __builtin_amdgcn_wmma_f32_16x16x32_bf16(false, af, false, b3, (short)0, acc3, false, false);
        }
        __syncthreads();
    }

    int n = lane & 15, mh = lane >> 4;
    v8f accs[4] = {acc0, acc1, acc2, acc3};
#pragma unroll
    for (int nt = 0; nt < 4; ++nt) {
        int col0 = n0 + nt * 16 + n;
        float bval = bias ? bias[col0] : 0.f;
#pragma unroll
        for (int r = 0; r < 8; ++r) {
            int m = m0 + r + 8 * mh;
            float v = accs[nt][r] + bval;
            if (gelu_flag) v = gelu_f(v);
            size_t idx = (size_t)m * ldc + col0;
            if (residual) v += residual[idx];
            if (outF) outF[idx] = v;
            if (outB) {
                if (kT_flag) {
                    // (row = b*1024+p, col = h*64+d) -> kT[((b*8+h)*64+d)*1024 + p]
                    size_t bb = (size_t)(m >> 10);
                    size_t p  = (size_t)(m & 1023);
                    size_t hh = (size_t)(col0 >> 6);
                    size_t dd = (size_t)(col0 & 63);
                    outB[((bb * HEADS_ + hh) * HD_ + dd) * NC_ + p] = (__bf16)v;
                } else {
                    outB[idx] = (__bf16)v;
                }
            }
        }
    }
}

// ---------------------------------------------------------------------------
// Focal modulation branch: per (b, c) plane. f layout: [B][Nc][NPAD] f32.
// conv3x3 -> gelu -> *gate0 ; conv5x5 -> gelu -> *gate1 ; gelu(mean) * gate2.
// Writes acc in [B][Nc][C] bf16 (pixel-major) ready to be GEMM A-matrix.
// ---------------------------------------------------------------------------
#define PW 36
__global__ void focal_kernel(const float* __restrict__ f, const float* __restrict__ fk0,
                             const float* __restrict__ fk1, __bf16* __restrict__ accb) {
    int bc = blockIdx.x;
    int b = bc >> 9, c = bc & 511;
    int tid = threadIdx.x;                  // 256 threads, 4 pixels each
    __shared__ float p0[PW * PW], p1[PW * PW];
    __shared__ float red[8];
    for (int i = tid; i < PW * PW; i += 256) { p0[i] = 0.f; p1[i] = 0.f; }
    __syncthreads();

    const float* fb = f + (size_t)b * NC_ * NPAD;
#pragma unroll
    for (int i = 0; i < 4; ++i) {
        int p = tid + 256 * i, py = p >> 5, px = p & 31;
        p0[(py + 2) * PW + (px + 2)] = fb[(size_t)p * NPAD + c];
    }
    __syncthreads();

    float w3[9];
#pragma unroll
    for (int j = 0; j < 9; ++j) w3[j] = fk0[c * 9 + j];
    float acc[4];
#pragma unroll
    for (int i = 0; i < 4; ++i) {
        int p = tid + 256 * i, py = p >> 5, px = p & 31;
        float s = 0.f;
#pragma unroll
        for (int ky = 0; ky < 3; ++ky)
#pragma unroll
            for (int kx = 0; kx < 3; ++kx)
                s += p0[(py + 1 + ky) * PW + (px + 1 + kx)] * w3[ky * 3 + kx];
        float v = gelu_f(s);
        p1[(py + 2) * PW + (px + 2)] = v;
        acc[i] = v * fb[(size_t)p * NPAD + 512];
    }
    __syncthreads();

    float w5[25];
#pragma unroll
    for (int j = 0; j < 25; ++j) w5[j] = fk1[c * 25 + j];
    float lsum = 0.f;
#pragma unroll
    for (int i = 0; i < 4; ++i) {
        int p = tid + 256 * i, py = p >> 5, px = p & 31;
        float s = 0.f;
#pragma unroll
        for (int ky = 0; ky < 5; ++ky)
#pragma unroll
            for (int kx = 0; kx < 5; ++kx)
                s += p1[(py + ky) * PW + (px + kx)] * w5[ky * 5 + kx];
        float v2 = gelu_f(s);
        lsum += v2;
        acc[i] += v2 * fb[(size_t)p * NPAD + 513];
    }
    for (int off = 16; off; off >>= 1) lsum += __shfl_xor(lsum, off);
    if ((tid & 31) == 0) red[tid >> 5] = lsum;
    __syncthreads();
    if (tid == 0) {
        float t = 0.f;
        for (int i = 0; i < 8; ++i) t += red[i];
        red[0] = t;
    }
    __syncthreads();
    float g = gelu_f(red[0] / (float)NC_);
#pragma unroll
    for (int i = 0; i < 4; ++i) {
        int p = tid + 256 * i;
        acc[i] += g * fb[(size_t)p * NPAD + 514];
        accb[((size_t)b * NC_ + p) * C_ + c] = (__bf16)acc[i];
    }
}

// ---------------------------------------------------------------------------
// Fused attention: one wave per (b, h, 16-row Q tile).
// S row-block (16 x 1024 f32) lives in dynamic LDS; full softmax; P @ V on WMMA.
// ---------------------------------------------------------------------------
__global__ void attn_kernel(const __bf16* __restrict__ q, const __bf16* __restrict__ kT,
                            const __bf16* __restrict__ v, __bf16* __restrict__ out) {
    extern __shared__ float smem[];
    float* S    = smem;              // 16 * 1024
    float* rsum = smem + 16 * 1024;  // 16
    int lane = threadIdx.x;
    int qt = blockIdx.x, h = blockIdx.y, b = blockIdx.z;
    const float scale = 0.125f;      // 64^-0.5

    const __bf16* qbase = q + ((size_t)(b * NL_ + qt * 16)) * C_ + h * HD_;
    v16bf qa0 = load_a_frag(qbase, C_, lane);
    v16bf qa1 = load_a_frag(qbase + 32, C_, lane);

    const __bf16* kbase = kT + ((size_t)(b * HEADS_ + h)) * HD_ * NC_;
    int n = lane & 15, mh = lane >> 4;
    for (int kt = 0; kt < NC_ / 16; ++kt) {
        v16bf kb0 = load_b_frag(kbase + (size_t)0 * NC_ + kt * 16, NC_, lane);
        v16bf kb1 = load_b_frag(kbase + (size_t)32 * NC_ + kt * 16, NC_, lane);
        v8f c = {};
        c = __builtin_amdgcn_wmma_f32_16x16x32_bf16(false, qa0, false, kb0, (short)0, c, false, false);
        c = __builtin_amdgcn_wmma_f32_16x16x32_bf16(false, qa1, false, kb1, (short)0, c, false, false);
#pragma unroll
        for (int r = 0; r < 8; ++r)
            S[(r + 8 * mh) * NC_ + kt * 16 + n] = c[r] * scale;
    }
    __syncthreads();

    for (int m = 0; m < 16; ++m) {
        float mx = -1e30f;
        for (int i = lane; i < NC_; i += 32) mx = fmaxf(mx, S[m * NC_ + i]);
        for (int off = 16; off; off >>= 1) mx = fmaxf(mx, __shfl_xor(mx, off));
        float sum = 0.f;
        for (int i = lane; i < NC_; i += 32) {
            float e = __expf(S[m * NC_ + i] - mx);
            S[m * NC_ + i] = e;
            sum += e;
        }
        for (int off = 16; off; off >>= 1) sum += __shfl_xor(sum, off);
        if (lane == 0) rsum[m] = sum;
    }
    __syncthreads();

    v8f o0 = {}, o1 = {}, o2 = {}, o3 = {};
    int am = lane & 15, ah = lane >> 4;
    for (int kt2 = 0; kt2 < NC_ / 32; ++kt2) {
        v16bf pa;
#pragma unroll
        for (int e = 0; e < 16; ++e) {
            int kk = 32 * kt2 + 8 * ah + e + (e >= 8 ? 8 : 0);
            pa[e] = (__bf16)S[am * NC_ + kk];
        }
        const __bf16* vb = v + ((size_t)(b * NC_ + 32 * kt2)) * C_ + h * HD_;
        v16bf b0 = load_b_frag(vb,      C_, lane);
        v16bf b1 = load_b_frag(vb + 16, C_, lane);
        v16bf b2 = load_b_frag(vb + 32, C_, lane);
        v16bf b3 = load_b_frag(vb + 48, C_, lane);
        o0 = __builtin_amdgcn_wmma_f32_16x16x32_bf16(false, pa, false, b0, (short)0, o0, false, false);
        o1 = __builtin_amdgcn_wmma_f32_16x16x32_bf16(false, pa, false, b1, (short)0, o1, false, false);
        o2 = __builtin_amdgcn_wmma_f32_16x16x32_bf16(false, pa, false, b2, (short)0, o2, false, false);
        o3 = __builtin_amdgcn_wmma_f32_16x16x32_bf16(false, pa, false, b3, (short)0, o3, false, false);
    }

#pragma unroll
    for (int r = 0; r < 8; ++r) {
        int m = r + 8 * mh;
        float inv = 1.0f / rsum[m];
        size_t base = ((size_t)(b * NL_ + qt * 16 + m)) * C_ + h * HD_ + n;
        out[base +  0] = (__bf16)(o0[r] * inv);
        out[base + 16] = (__bf16)(o1[r] * inv);
        out[base + 32] = (__bf16)(o2[r] * inv);
        out[base + 48] = (__bf16)(o3[r] * inv);
    }
}

// ---------------------------------------------------------------------------
// Host orchestration
// ---------------------------------------------------------------------------
extern "C" void kernel_launch(void* const* d_in, const int* in_sizes, int n_in,
                              void* d_out, int out_size, void* d_ws, size_t ws_size,
                              hipStream_t stream) {
    const float* latents = (const float*)d_in[0];
    const float* context = (const float*)d_in[1];
    const float* norm_l_w = (const float*)d_in[4];
    const float* norm_l_b = (const float*)d_in[5];
    const float* norm_c_w = (const float*)d_in[6];
    const float* norm_c_b = (const float*)d_in[7];
    const float* norm2_w  = (const float*)d_in[8];
    const float* norm2_b  = (const float*)d_in[9];
    const float* Wq = (const float*)d_in[10];
    const float* Wk = (const float*)d_in[11];
    const float* Wf = (const float*)d_in[12];
    const float* bfv = (const float*)d_in[13];
    const float* fk0 = (const float*)d_in[14];
    const float* fk1 = (const float*)d_in[15];
    const float* h_w = (const float*)d_in[16];
    const float* h_b = (const float*)d_in[17];
    const float* Wproj = (const float*)d_in[18];
    const float* bproj = (const float*)d_in[19];
    const float* fc1_w = (const float*)d_in[20];
    const float* fc1_bias = (const float*)d_in[21];
    const float* fc2_w = (const float*)d_in[22];
    const float* fc2_bias = (const float*)d_in[23];
    float* outp = (float*)d_out;

    const int Mq = B_ * NL_;   // 8192
    const int Mc = B_ * NC_;   // 16384

    char* base = (char*)d_ws;
    size_t off = 0;
    auto alloc = [&](size_t bytes) -> void* {
        void* r = base + off;
        off = (off + bytes + 255) & ~(size_t)255;
        return r;
    };
    __bf16* xlb  = (__bf16*)alloc((size_t)Mq * C_ * 2);
    __bf16* xcb  = (__bf16*)alloc((size_t)Mc * C_ * 2);
    __bf16* wqb  = (__bf16*)alloc((size_t)C_ * C_ * 2);
    __bf16* wkb  = (__bf16*)alloc((size_t)C_ * C_ * 2);
    __bf16* wpb  = (__bf16*)alloc((size_t)C_ * C_ * 2);
    __bf16* hwb  = (__bf16*)alloc((size_t)C_ * C_ * 2);
    __bf16* wfb  = (__bf16*)alloc((size_t)C_ * NPAD * 2);
    float*  bfp  = (float*) alloc((size_t)NPAD * 4);
    __bf16* w1b  = (__bf16*)alloc((size_t)C_ * 2048 * 2);
    __bf16* w2b  = (__bf16*)alloc((size_t)2048 * C_ * 2);
    __bf16* qb   = (__bf16*)alloc((size_t)Mq * C_ * 2);
    __bf16* kTb  = (__bf16*)alloc((size_t)B_ * HEADS_ * HD_ * NC_ * 2);
    float*  ff   = (float*) alloc((size_t)Mc * NPAD * 4);   // dies after focal; m1b aliases
    __bf16* accb = (__bf16*)alloc((size_t)Mc * C_ * 2);
    __bf16* vb   = (__bf16*)alloc((size_t)Mc * C_ * 2);
    float*  x1   = (float*) alloc((size_t)Mq * C_ * 4);
    __bf16* h2b  = (__bf16*)alloc((size_t)Mq * C_ * 2);
    __bf16* attb = xlb;             // xl dead after q GEMM
    __bf16* m1b  = (__bf16*)ff;     // f dead after focal (32 MB <= 37.7 MB)
    (void)ws_size; (void)in_sizes; (void)n_in; (void)out_size;

    // 1) LayerNorms -> bf16
    ln_kernel<<<Mq, 128, 0, stream>>>(latents, norm_l_w, norm_l_b, xlb, C_);
    ln_kernel<<<Mc, 128, 0, stream>>>(context, norm_c_w, norm_c_b, xcb, C_);

    // 2) weight conversions
    auto cv = [&](const float* s, __bf16* d, int n) {
        cvt_kernel<<<(n + 255) / 256, 256, 0, stream>>>(s, d, n);
    };
    cv(Wq, wqb, C_ * C_); cv(Wk, wkb, C_ * C_);
    cv(Wproj, wpb, C_ * C_); cv(h_w, hwb, C_ * C_);
    cv(fc1_w, w1b, C_ * 2048); cv(fc2_w, w2b, 2048 * C_);
    pad_wf_kernel<<<(C_ * NPAD + 255) / 256, 256, 0, stream>>>(Wf, wfb);
    pad_bias_kernel<<<3, 256, 0, stream>>>(bfv, bfp);

    // 3) q = xl @ Wq  (bf16 out)
    gemm_bf16_kernel<<<dim3(C_ / 64, Mq / 64), 128, 0, stream>>>(
        xlb, wqb, nullptr, nullptr, nullptr, qb, Mq, C_, C_, C_, C_, C_, 0, 0);
    // 4) k = xc @ Wk, scattered into kT[B][H][hd][Nc]
    gemm_bf16_kernel<<<dim3(C_ / 64, Mc / 64), 128, 0, stream>>>(
        xcb, wkb, nullptr, nullptr, nullptr, kTb, Mc, C_, C_, C_, C_, C_, 0, 1);
    // 5) f = xc @ Wf + bf  (f32, padded N)
    gemm_bf16_kernel<<<dim3(NPAD / 64, Mc / 64), 128, 0, stream>>>(
        xcb, wfb, bfp, nullptr, ff, nullptr, Mc, NPAD, C_, C_, NPAD, NPAD, 0, 0);
    // 6) focal modulation -> acc bf16 [B][Nc][C]
    focal_kernel<<<B_ * C_, 256, 0, stream>>>(ff, fk0, fk1, accb);
    // 7) v = acc @ h_w + h_b  (bf16)
    gemm_bf16_kernel<<<dim3(C_ / 64, Mc / 64), 128, 0, stream>>>(
        accb, hwb, h_b, nullptr, nullptr, vb, Mc, C_, C_, C_, C_, C_, 0, 0);
    // 8) fused attention
    attn_kernel<<<dim3(NL_ / 16, HEADS_, B_), 32, 16 * NC_ * 4 + 64, stream>>>(
        qb, kTb, vb, attb);
    // 9) x1 = latents + attn @ Wproj + bproj  (f32)
    gemm_bf16_kernel<<<dim3(C_ / 64, Mq / 64), 128, 0, stream>>>(
        attb, wpb, bproj, latents, x1, nullptr, Mq, C_, C_, C_, C_, C_, 0, 0);
    // 10) h2 = LN(x1)
    ln_kernel<<<Mq, 128, 0, stream>>>(x1, norm2_w, norm2_b, h2b, C_);
    // 11) m1 = gelu(h2 @ fc1 + b)  (bf16)
    gemm_bf16_kernel<<<dim3(2048 / 64, Mq / 64), 128, 0, stream>>>(
        h2b, w1b, fc1_bias, nullptr, nullptr, m1b, Mq, 2048, C_, C_, 2048, 2048, 1, 0);
    // 12) out = x1 + m1 @ fc2 + b  (f32)
    gemm_bf16_kernel<<<dim3(C_ / 64, Mq / 64), 128, 0, stream>>>(
        m1b, w2b, fc2_bias, x1, outp, nullptr, Mq, C_, 2048, 2048, C_, C_, 0, 0);
}
